// TranswaveUNET_c_53996328845920
// MI455X (gfx1250) — matrix operand order repaired and checked by reading, not compiled, not run on hardware
//
#include <hip/hip_runtime.h>
#include <math.h>
#include <stdint.h>

// ---------------- CDNA5 feature detection (compile-safe on both toolchains) ----
#define ASG __attribute__((address_space(1)))
#define ASL __attribute__((address_space(3)))

#if defined(__gfx1250__) && __has_builtin(__builtin_amdgcn_global_load_async_to_lds_b32) && \
    __has_builtin(__builtin_amdgcn_s_wait_asynccnt)
#define USE_ASYNC_LDS 1
#else
#define USE_ASYNC_LDS 0
#endif

#if defined(__gfx1250__) && __has_builtin(__builtin_amdgcn_tensor_load_to_lds) && \
    __has_builtin(__builtin_amdgcn_s_wait_tensorcnt)
#define USE_TDM 1
#else
#define USE_TDM 0
#endif

typedef unsigned int uint32x4 __attribute__((ext_vector_type(4)));
typedef int          int32x4  __attribute__((ext_vector_type(4)));
typedef int          int32x8  __attribute__((ext_vector_type(8)));

// ---------------- CDNA5 WMMA primitives (wave32) ----------------
typedef __attribute__((ext_vector_type(16))) __bf16 v16bf;
typedef __attribute__((ext_vector_type(8)))  float  v8f;

__device__ __forceinline__ __bf16 f2bf(float f) {
  union { float f; unsigned u; } x; x.f = f;
  unsigned r = x.u + 0x7FFFu + ((x.u >> 16) & 1u);  // RNE
  union { unsigned short s; __bf16 b; } y; y.s = (unsigned short)(r >> 16);
  return y.b;
}

__device__ __forceinline__ v8f wmma_bf16(v16bf a, v16bf b, v8f c) {
  // D = A(16x32 bf16) * B(32x16 bf16) + C(16x16 f32)
  return __builtin_amdgcn_wmma_f32_16x16x32_bf16(false, a, false, b, (short)0, c,
                                                 false, false);
}

// async per-lane 4-byte copy global -> LDS (ASYNCcnt tracked)
// builtin signature: (int addrspace(1)*, int addrspace(3)*, imm offset, imm cpol)
#if USE_ASYNC_LDS
#define ASYNC_CP32(gp, lp)                                              \
  __builtin_amdgcn_global_load_async_to_lds_b32(                        \
      (ASG int*)(uintptr_t)(gp), (ASL int*)(uintptr_t)(lp), 0, 0)
#endif

// ---------------- Generic WMMA GEMM ----------------
// C[M,N] = act( A[M,K] * B[K,N] + bias + residual )
// a_mode: 0 plain A[m*lda+k]; 1 im2col from NCHW; 2 tokens from CHW (64 tok/batch)
// b_trans: 0 B[k*ldb+n] (TDM path); 1 B[n*ldb+k] (OIHW conv weights)
// c_mode: 0 plain C[m*N+n]; 1 scatter to NCHW
// res_mode: 0 none; 1 R[m*N+n]; 2 R[(m%64)*N+n] (positional embedding)
// act: 0 none, 1 relu, 2 gelu(tanh)
struct ConvDesc { int IC, IH, IW, KH, KW, stride, pad, OH, OW; };

#define BM 128
#define BN 64
#define BK 32

__global__ __launch_bounds__(256) void gemm_bf16_k(
    const float* __restrict__ A, const float* __restrict__ Bm,
    const float* __restrict__ bias, const float* __restrict__ R,
    float* __restrict__ C, const float* __restrict__ zp,
    int M, int N, int K, int lda, int ldb,
    int a_mode, int b_trans, int c_mode, int res_mode, int act,
    ConvDesc cd)
{
  __shared__ float Ash[BM * 33];   // [m*33 + k]
  __shared__ float Bsh[BN * 33];   // async: [n*33+k] ; TDM tile: [k*64+n]

  const int tid  = threadIdx.x;
  const int wv   = tid >> 5;          // 8 waves
  const int lane = tid & 31;
  const int lm   = lane & 15;
  const int half = lane >> 4;         // 0 | 1
  const int wr   = wv & 3;            // wave row (4)
  const int wc   = wv >> 2;           // wave col (2)
  const int m0   = blockIdx.y * BM;
  const int n0   = blockIdx.x * BN;

#if USE_TDM
  const bool tdm_b = (b_trans == 0);  // NN GEMMs stage B via Tensor Data Mover
#else
  const bool tdm_b = false;
#endif

  v8f acc[2][2];
  #pragma unroll
  for (int i = 0; i < 2; ++i)
    #pragma unroll
    for (int j = 0; j < 2; ++j)
      acc[i][j] = {};

  const int khkw = cd.KH * cd.KW;
  const int ohow = cd.OH * cd.OW;

  for (int k0 = 0; k0 < K; k0 += BK) {
    __syncthreads();

    if (a_mode == 0 && k0 + BK < K) {
      int pm = m0 + (tid & (BM - 1));
      if (pm < M) __builtin_prefetch(A + (size_t)pm * lda + k0 + BK, 0, 0);
    }

    // ---- stage A tile (128x32 f32), per-lane async gather ----
    for (int t = tid; t < BM * BK; t += 256) {
      int r = t >> 5, kk = t & 31;
      int gm = m0 + r, gk = k0 + kk;
      const float* src = zp;            // zero page covers all padding cases
      if (gm < M && gk < K) {
        if (a_mode == 0) {
          src = A + (size_t)gm * lda + gk;
        } else if (a_mode == 2) {
          src = A + ((size_t)(gm >> 6) * K + gk) * 64 + (gm & 63);
        } else { // im2col from NCHW
          int ic = gk / khkw, r2 = gk - ic * khkw;
          int kh = r2 / cd.KW, kw = r2 - kh * cd.KW;
          int bb = gm / ohow, r3 = gm - bb * ohow;
          int oh = r3 / cd.OW, ow = r3 - oh * cd.OW;
          int ih = oh * cd.stride + kh - cd.pad;
          int iw = ow * cd.stride + kw - cd.pad;
          if (ih >= 0 && ih < cd.IH && iw >= 0 && iw < cd.IW)
            src = A + (((size_t)bb * cd.IC + ic) * cd.IH + ih) * cd.IW + iw;
        }
      }
#if USE_ASYNC_LDS
      ASYNC_CP32(src, &Ash[r * 33 + kk]);
#else
      Ash[r * 33 + kk] = *src;
#endif
    }

    // ---- stage B tile (32x64 f32) ----
    if (tdm_b) {
#if USE_TDM
      if (tid < 32) {  // one wave issues the TDM descriptor
        unsigned lds_base = (unsigned)(uintptr_t)&Bsh[0];
        unsigned long long ga =
            (unsigned long long)(uintptr_t)(Bm + (size_t)k0 * ldb + n0);
        unsigned td0 = (unsigned)(N - n0);   // remaining cols (dim0, contiguous)
        unsigned td1 = (unsigned)(K - k0);   // remaining rows (dim1)
        unsigned long long st0 = (unsigned long long)ldb;
        uint32x4 g0;
        g0[0] = 1u;                                   // count=1, user descriptor
        g0[1] = lds_base;                             // lds_addr
        g0[2] = (unsigned)(ga & 0xFFFFFFFFull);       // global_addr[31:0]
        g0[3] = (unsigned)((ga >> 32) & 0x1FFFFFFull) // global_addr[56:32]
                | 0x80000000u;                        // type=2 ("image")
        int32x8 g1;
        g1[0] = (int)(2u << 16);                      // data_size = 4 bytes
        g1[1] = (int)((td0 & 0xFFFFu) << 16);         // tensor_dim0[15:0]
        g1[2] = (int)((td0 >> 16) | ((td1 & 0xFFFFu) << 16));
        g1[3] = (int)((td1 >> 16) | (64u << 16));     // tile_dim0 = 64
        g1[4] = (int)32u;                             // tile_dim1 = 32, tile_dim2 = 0
        g1[5] = (int)(unsigned)(st0 & 0xFFFFFFFFull); // tensor_dim0_stride lo
        g1[6] = (int)(unsigned)((st0 >> 32) & 0xFFFFull);
        g1[7] = 0;
        int32x4 gz = {0, 0, 0, 0};
#if __clang_major__ >= 23
        int32x8 gz8 = {0, 0, 0, 0, 0, 0, 0, 0};
        __builtin_amdgcn_tensor_load_to_lds(g0, g1, gz, gz, gz8, 0);
#else
        __builtin_amdgcn_tensor_load_to_lds(g0, g1, gz, gz, 0);
#endif
      }
#endif
    } else {
      for (int t = tid; t < BN * BK; t += 256) {
        int nn = t >> 5, kk = t & 31;
        int gn = n0 + nn, gk = k0 + kk;
        const float* src = zp;
        if (gn < N && gk < K)
          src = b_trans ? Bm + (size_t)gn * ldb + gk
                        : Bm + (size_t)gk * ldb + gn;
#if USE_ASYNC_LDS
        ASYNC_CP32(src, &Bsh[nn * 33 + kk]);
#else
        Bsh[nn * 33 + kk] = *src;
#endif
      }
    }

#if USE_ASYNC_LDS
    __builtin_amdgcn_s_wait_asynccnt(0);
#endif
#if USE_TDM
    if (tdm_b) __builtin_amdgcn_s_wait_tensorcnt(0);
#endif
    __syncthreads();

    // ---- fragments in contiguous k order (k = 16*half + e) for A and B (f32 LDS -> bf16 regs) ----
    v16bf af[2], bfv[2];
    #pragma unroll
    for (int i = 0; i < 2; ++i) {
      int row = wr * 32 + i * 16 + lm;
      int kb  = half * 16;
      #pragma unroll
      for (int e = 0; e < 16; ++e)
        af[i][e] = f2bf(Ash[row * 33 + kb + e]);
    }
    const int bstep = tdm_b ? 64 : 1;   // uniform select, branch-free frag build
    #pragma unroll
    for (int j = 0; j < 2; ++j) {
      int col = wc * 32 + j * 16 + lm;
      int kb  = half * 16;
      int base = tdm_b ? (kb * 64 + col) : (col * 33 + kb);
      #pragma unroll
      for (int e = 0; e < 16; ++e)
        bfv[j][e] = f2bf(Bsh[base + e * bstep]);
    }
    #pragma unroll
    for (int i = 0; i < 2; ++i)
      #pragma unroll
      for (int j = 0; j < 2; ++j)
        acc[i][j] = wmma_bf16(af[i], bfv[j], acc[i][j]);
  }

  // ---- epilogue ----
  #pragma unroll
  for (int i = 0; i < 2; ++i) {
    #pragma unroll
    for (int j = 0; j < 2; ++j) {
      int colbase = n0 + wc * 32 + j * 16 + lm;
      #pragma unroll
      for (int e = 0; e < 8; ++e) {
        int row = m0 + wr * 32 + i * 16 + half * 8 + e;
        int col = colbase;
        if (row < M && col < N) {
          float v = acc[i][j][e];
          if (bias) v += bias[col];
          if (res_mode == 1) v += R[(size_t)row * N + col];
          else if (res_mode == 2) v += R[(size_t)(row & 63) * N + col];
          if (act == 1) v = fmaxf(v, 0.f);
          else if (act == 2) {
            float u = v;
            v = 0.5f * u * (1.f + tanhf(0.7978845608f * (u + 0.044715f * u * u * u)));
          }
          if (c_mode == 0) C[(size_t)row * N + col] = v;
          else {
            int bb = row / ohow, hw = row - bb * ohow;
            C[((size_t)bb * N + col) * ohow + hw] = v;
          }
        }
      }
    }
  }
}

// ---------------- Fused attention per (batch, head): N=64 tok, Dh=256 ----------------
__global__ __launch_bounds__(256) void attn_k(const float* __restrict__ qkv,
                                              float* __restrict__ o)
{
  const int bh = blockIdx.x, b = bh >> 2, h = bh & 3;
  __shared__ __bf16 Qs[64][256];
  __shared__ float  Ss[64][65];
  __shared__ __bf16 Ps[64][64];

  const int tid = threadIdx.x;
  const int wv = tid >> 5, lane = tid & 31, lm = lane & 15, half = lane >> 4;
  const float* qbase = qkv + (size_t)b * 64 * 3072 + h * 256;
  const float* kbase = qbase + 1024;
  const float* vbase = qbase + 2048;

  for (int t = tid; t < 64 * 256; t += 256) {
    int n = t >> 8, d = t & 255;
    Qs[n][d] = f2bf(qbase[(size_t)n * 3072 + d]);
  }
  __syncthreads();

  // S = (Q K^T) * 1/sqrt(256); 16 tiles of 16x16, 2 per wave
  for (int t = 0; t < 2; ++t) {
    int tile = wv * 2 + t;
    int r0 = (tile >> 2) * 16, c0 = (tile & 3) * 16;
    v8f acc = {};
    for (int kk = 0; kk < 256; kk += 32) {
      v16bf af, bf_;
      int kb = half * 16;
      #pragma unroll
      for (int e = 0; e < 16; ++e)
        af[e] = Qs[r0 + lm][kk + kb + e];
      const float* kr = kbase + (size_t)(c0 + lm) * 3072 + kk + half * 16;
      #pragma unroll
      for (int e = 0; e < 16; ++e) bf_[e] = f2bf(kr[e]);   // NT: B[k][n]=K[n][k]
      acc = wmma_bf16(af, bf_, acc);
    }
    #pragma unroll
    for (int e = 0; e < 8; ++e)
      Ss[r0 + half * 8 + e][c0 + lm] = acc[e] * 0.0625f;
  }
  __syncthreads();

  if (tid < 64) {   // softmax over row tid
    float mx = -1e30f;
    for (int j = 0; j < 64; ++j) mx = fmaxf(mx, Ss[tid][j]);
    float sm = 0.f;
    for (int j = 0; j < 64; ++j) { float e = expf(Ss[tid][j] - mx); Ss[tid][j] = e; sm += e; }
    float inv = 1.f / sm;
    for (int j = 0; j < 64; ++j) Ps[tid][j] = f2bf(Ss[tid][j] * inv);
  }
  __syncthreads();

  // O = P V ; 64 tiles (4 x 16), 8 per wave
  float* obase = o + (size_t)b * 64 * 1024 + h * 256;
  for (int t = 0; t < 8; ++t) {
    int tile = wv * 8 + t;
    int r0 = (tile >> 4) * 16, c0 = (tile & 15) * 16;
    v8f acc = {};
    for (int kk = 0; kk < 64; kk += 32) {
      v16bf af, bf_;
      int kb = half * 16;
      #pragma unroll
      for (int e = 0; e < 16; ++e)
        af[e] = Ps[r0 + lm][kk + kb + e];
      #pragma unroll
      for (int e = 0; e < 16; ++e) {
        int krow = kk + half * 16 + e;
        bf_[e] = f2bf(vbase[(size_t)krow * 3072 + c0 + lm]);
      }
      acc = wmma_bf16(af, bf_, acc);
    }
    #pragma unroll
    for (int e = 0; e < 8; ++e)
      obase[(size_t)(r0 + half * 8 + e) * 1024 + c0 + lm] = acc[e];
  }
}

// ---------------- Elementwise / reduction helpers ----------------
__global__ void zerofill_k(float* __restrict__ p, int n) {
  int i = blockIdx.x * 256 + threadIdx.x;
  if (i < n) p[i] = 0.f;
}

__global__ void bn_stats_k(const float* __restrict__ x, float* __restrict__ st,
                           int C, int HW, int B) {
  int c = blockIdx.x, tid = threadIdx.x;
  float s = 0.f, q = 0.f;
  for (int b = 0; b < B; ++b) {
    const float* p = x + ((size_t)b * C + c) * HW;
    for (int i = tid; i < HW; i += 256) { float v = p[i]; s += v; q += v * v; }
  }
  __shared__ float rs[256], rq[256];
  rs[tid] = s; rq[tid] = q; __syncthreads();
  for (int st2 = 128; st2 > 0; st2 >>= 1) {
    if (tid < st2) { rs[tid] += rs[tid + st2]; rq[tid] += rq[tid + st2]; }
    __syncthreads();
  }
  if (tid == 0) { st[c] = rs[0]; st[C + c] = rq[0]; }
}

__global__ void bn_apply_relu_k(float* __restrict__ x, const float* __restrict__ st,
                                const float* __restrict__ g, const float* __restrict__ bt,
                                int C, int HW, int B) {
  size_t idx = (size_t)blockIdx.x * 256 + threadIdx.x;
  size_t total = (size_t)B * C * HW;
  if (idx >= total) return;
  int c = (int)((idx / (size_t)HW) % C);
  float cnt = (float)B * (float)HW;
  float m = st[c] / cnt;
  float var = st[C + c] / cnt - m * m;
  float v = (x[idx] - m) * rsqrtf(var + 1e-5f) * g[c] + bt[c];
  x[idx] = v > 0.f ? v : 0.f;
}

__global__ void layernorm_k(const float* __restrict__ x, float* __restrict__ y,
                            const float* __restrict__ g, const float* __restrict__ b,
                            int D) {
  int row = blockIdx.x, tid = threadIdx.x;
  const float* p = x + (size_t)row * D;
  float s = 0.f, q = 0.f;
  for (int i = tid; i < D; i += 256) { float v = p[i]; s += v; q += v * v; }
  __shared__ float rs[256], rq[256];
  rs[tid] = s; rq[tid] = q; __syncthreads();
  for (int st = 128; st > 0; st >>= 1) {
    if (tid < st) { rs[tid] += rs[tid + st]; rq[tid] += rq[tid + st]; }
    __syncthreads();
  }
  __shared__ float mean, inv;
  if (tid == 0) {
    mean = rs[0] / D;
    float var = rq[0] / D - mean * mean;
    inv = rsqrtf(var + 1e-5f);
  }
  __syncthreads();
  for (int i = tid; i < D; i += 256)
    y[(size_t)row * D + i] = (p[i] - mean) * inv * g[i] + b[i];
}

__global__ void dwt_k(const float* __restrict__ in, float* __restrict__ out,
                      int B, int C, int H, int W) {
  int h2 = H >> 1, w2 = W >> 1;
  size_t idx = (size_t)blockIdx.x * 256 + threadIdx.x;
  size_t total = (size_t)B * C * h2 * w2;
  if (idx >= total) return;
  int j = (int)(idx % w2); size_t r = idx / w2;
  int i = (int)(r % h2); r /= h2;
  int c = (int)(r % C);  int b = (int)(r / C);
  const float* p = in + ((size_t)b * C + c) * H * W;
  float a  = p[(2 * i) * W + 2 * j],     bb = p[(2 * i) * W + 2 * j + 1];
  float cc = p[(2 * i + 1) * W + 2 * j], dd = p[(2 * i + 1) * W + 2 * j + 1];
  size_t chs = (size_t)C * h2 * w2;
  size_t o = (((size_t)b * 4 * C) + c) * h2 * w2 + (size_t)i * w2 + j;
  out[o]            = (a + bb + cc + dd) * 0.5f;
  out[o + chs]      = (a + bb - cc - dd) * 0.5f;
  out[o + 2 * chs]  = (a - bb + cc - dd) * 0.5f;
  out[o + 3 * chs]  = (a - bb - cc + dd) * 0.5f;
}

__global__ void idwt_k(const float* __restrict__ cA, const float* __restrict__ hp,
                       float* __restrict__ out, int B, int C, int h, int w, int Ctot) {
  size_t idx = (size_t)blockIdx.x * 256 + threadIdx.x;
  size_t total = (size_t)B * C * h * w;
  if (idx >= total) return;
  int j = (int)(idx % w); size_t r = idx / w;
  int i = (int)(r % h); r /= h;
  int c = (int)(r % C); int b = (int)(r / C);
  float A = cA[idx];
  float Hh = hp ? hp[idx] : 0.f;
  float a = (A + 3.f * Hh) * 0.5f, bcd = (A - Hh) * 0.5f;
  float* o = out + (((size_t)b * Ctot + c) * (2 * h) + 2 * i) * (size_t)(2 * w) + 2 * j;
  o[0] = a; o[1] = bcd; o[2 * w] = bcd; o[2 * w + 1] = bcd;
}

__global__ void copych_k(const float* __restrict__ src, float* __restrict__ dst,
                         int B, int C, int HW, int Ctot, int coff) {
  size_t idx = (size_t)blockIdx.x * 256 + threadIdx.x;
  size_t total = (size_t)B * C * HW;
  if (idx >= total) return;
  int p = (int)(idx % HW); size_t r = idx / HW;
  int c = (int)(r % C); int b = (int)(r / C);
  dst[((size_t)b * Ctot + coff + c) * HW + p] = src[idx];
}

__global__ void tok2nchw_k(const float* __restrict__ tok, float* __restrict__ out,
                           int B, int Np, int C) {
  size_t idx = (size_t)blockIdx.x * 256 + threadIdx.x;
  size_t total = (size_t)B * Np * C;
  if (idx >= total) return;
  int c = (int)(idx % C); size_t r = idx / C;
  int n = (int)(r % Np); int b = (int)(r / Np);
  out[((size_t)b * C + c) * Np + n] = tok[idx];
}

__global__ void outconv_k(const float* __restrict__ x, const float* __restrict__ w,
                          const float* __restrict__ bias, float* __restrict__ out,
                          int B, int IC, int HW) {
  size_t idx = (size_t)blockIdx.x * 256 + threadIdx.x;
  if (idx >= (size_t)B * HW) return;
  int p = (int)(idx % HW), b = (int)(idx / HW);
  float s = bias[0];
  for (int ic = 0; ic < IC; ++ic)
    s += x[((size_t)b * IC + ic) * HW + p] * w[ic];
  out[idx] = s;
}

// ---------------- Host orchestration ----------------
struct CBN { const float *b, *beta, *g, *w; };

static inline unsigned cdiv(size_t a, size_t b) { return (unsigned)((a + b - 1) / b); }

static void gemm(hipStream_t s, const float* A, const float* Bm, const float* bias,
                 const float* R, float* C, const float* zp,
                 int M, int N, int K, int lda, int ldb,
                 int a_mode, int b_trans, int c_mode, int res_mode, int act,
                 const ConvDesc& cd) {
  dim3 grid(cdiv(N, BN), cdiv(M, BM), 1);
  gemm_bf16_k<<<grid, 256, 0, s>>>(A, Bm, bias, R, C, zp, M, N, K, lda, ldb,
                                   a_mode, b_trans, c_mode, res_mode, act, cd);
}

static void conv_nchw(hipStream_t s, const float* x, const float* w, const float* bias,
                      float* y, const float* zp, int B, int IC, int IH, int IW, int OC,
                      int k, int stride, int pad) {
  int OH = (IH + 2 * pad - k) / stride + 1;
  int OW = (IW + 2 * pad - k) / stride + 1;
  ConvDesc cd = {IC, IH, IW, k, k, stride, pad, OH, OW};
  gemm(s, x, w, bias, nullptr, y, zp, B * OH * OW, OC, IC * k * k,
       0, IC * k * k, /*a_mode=*/1, /*b_trans=*/1, /*c_mode=*/1, 0, 0, cd);
}

// conv (bias dropped: BN-invariant) + batchnorm(train stats) + relu, in-place BN
static void cbr(hipStream_t s, const float* x, const CBN& p, float* y, float* stats,
                const float* zp, int B, int IC, int IH, int IW, int OC, int k, int stride) {
  int pad = (k - 1) / 2;
  conv_nchw(s, x, p.w, nullptr, y, zp, B, IC, IH, IW, OC, k, stride, pad);
  int OH = (IH + 2 * pad - k) / stride + 1;
  int OW = (IW + 2 * pad - k) / stride + 1;
  bn_stats_k<<<OC, 256, 0, s>>>(y, stats, OC, OH * OW, B);
  size_t tot = (size_t)B * OC * OH * OW;
  bn_apply_relu_k<<<cdiv(tot, 256), 256, 0, s>>>(y, stats, p.g, p.beta, OC, OH * OW, B);
}

extern "C" void kernel_launch(void* const* d_in, const int* in_sizes, int n_in,
                              void* d_out, int out_size, void* d_ws, size_t ws_size,
                              hipStream_t stream) {
  (void)in_sizes; (void)n_in; (void)out_size; (void)ws_size;
  const int B = 16;
  ConvDesc cd0 = {};

  // ---- param pointers (jax pytree order: dict keys sorted alphabetically) ----
  int ii = 0;
  auto F = [&](void) -> const float* { return (const float*)d_in[ii++]; };
  const float* X = F();                              // inputs
  auto rdCBN = [&](void) { CBN c; c.b = F(); c.beta = F(); c.g = F(); c.w = F(); return c; };
  CBN bottle = rdCBN();                              // bottle: b,beta,g,w
  struct Down { CBN c1, c2, c3; const float *dwt_b, *dwt_w; };
  auto rdDown = [&](void) { Down d; d.c1 = rdCBN(); d.c2 = rdCBN(); d.c3 = rdCBN();
                            d.dwt_b = F(); d.dwt_w = F(); return d; };
  Down dn0 = rdDown(), dn1 = rdDown(), dn2 = rdDown();
  CBN first = rdCBN();
  const float* out_b = F(); const float* out_w = F();
  struct Up { CBN conv, dwt_up; };
  auto rdUp = [&](void) { Up u; u.conv = rdCBN(); u.dwt_up = rdCBN(); return u; };
  Up up0 = rdUp(), up1 = rdUp(), up2 = rdUp();
  CBN up3_last = rdCBN();
  struct Blk { const float *b1,*b2,*bo,*bqkv,*ln1b,*ln1g,*ln2b,*ln2g,*w1,*w2,*wo,*wqkv; };
  Blk blk[8];
  for (int i = 0; i < 8; ++i) {
    blk[i].b1 = F(); blk[i].b2 = F(); blk[i].bo = F(); blk[i].bqkv = F();
    blk[i].ln1b = F(); blk[i].ln1g = F(); blk[i].ln2b = F(); blk[i].ln2g = F();
    blk[i].w1 = F(); blk[i].w2 = F(); blk[i].wo = F(); blk[i].wqkv = F();
  }
  const float* embed_b = F(); const float* embed_w = F();
  const float* lnfb = F(); const float* lnfg = F(); const float* pos = F();

  // ---- workspace bump allocator ----
  size_t off = 0;
  auto alloc = [&](size_t elems) {
    float* p = (float*)((char*)d_ws + off);
    off += ((elems * sizeof(float) + 255) / 256) * 256;
    return p;
  };
  float* zp    = alloc(256);       // zero page for async-gather padding
  float* stats = alloc(2048);
  float* s0    = alloc((size_t)B * 128 * 64 * 64);
  float* t1    = alloc((size_t)B * 256 * 64 * 64);
  float* t2    = alloc((size_t)B * 256 * 32 * 32);
  float* s1    = alloc((size_t)B * 256 * 32 * 32);
  float* dc0   = alloc((size_t)B * 64 * 64 * 64);
  float* coef1 = alloc((size_t)B * 256 * 32 * 32);
  float* u1    = alloc((size_t)B * 512 * 32 * 32);
  float* u2    = alloc((size_t)B * 512 * 16 * 16);
  float* s2    = alloc((size_t)B * 512 * 16 * 16);
  float* dc1   = alloc((size_t)B * 128 * 32 * 32);
  float* coef2 = alloc((size_t)B * 512 * 16 * 16);
  float* v1    = alloc((size_t)B * 1024 * 16 * 16);
  float* v2    = alloc((size_t)B * 1024 * 8 * 8);
  float* dc2   = alloc((size_t)B * 256 * 16 * 16);
  float* coef3 = alloc((size_t)B * 1024 * 8 * 8);
  float* tokA  = alloc((size_t)B * 64 * 1024);
  float* tokB  = alloc((size_t)B * 64 * 1024);
  float* hbuf  = alloc((size_t)B * 64 * 1024);
  float* qkvb  = alloc((size_t)B * 64 * 3072);
  float* obuf  = alloc((size_t)B * 64 * 1024);
  float* m1b   = alloc((size_t)B * 64 * 512);
  float* xn    = alloc((size_t)B * 1024 * 64);
  float* bot   = alloc((size_t)B * 512 * 8 * 8);
  float* hp0   = alloc((size_t)B * 512 * 8 * 8);
  float* cat0  = alloc((size_t)B * 1024 * 16 * 16);
  float* i1    = alloc((size_t)B * 256 * 16 * 16);
  float* hp1   = alloc((size_t)B * 256 * 16 * 16);
  float* cat1  = alloc((size_t)B * 512 * 32 * 32);
  float* i2    = alloc((size_t)B * 128 * 32 * 32);
  float* hp2   = alloc((size_t)B * 128 * 32 * 32);
  float* cat2  = alloc((size_t)B * 256 * 64 * 64);
  float* i3    = alloc((size_t)B * 64 * 64 * 64);
  float* d4in  = alloc((size_t)B * 64 * 128 * 128);
  float* d4    = alloc((size_t)B * 16 * 128 * 128);

  hipStream_t s = stream;
  zerofill_k<<<1, 256, 0, s>>>(zp, 256);

  // ---- encoder ----
  cbr(s, X, first, s0, stats, zp, B, 3, 128, 128, 128, 3, 2);             // (B,128,64,64)

  cbr(s, s0, dn0.c1, t1, stats, zp, B, 128, 64, 64, 256, 1, 1);
  cbr(s, t1, dn0.c2, t2, stats, zp, B, 256, 64, 64, 256, 3, 2);           // 32x32
  cbr(s, t2, dn0.c3, s1, stats, zp, B, 256, 32, 32, 256, 1, 1);
  conv_nchw(s, s0, dn0.dwt_w, dn0.dwt_b, dc0, zp, B, 128, 64, 64, 64, 3, 1, 1);
  dwt_k<<<cdiv((size_t)B * 64 * 32 * 32, 256), 256, 0, s>>>(dc0, coef1, B, 64, 64, 64);

  cbr(s, coef1, dn1.c1, u1, stats, zp, B, 256, 32, 32, 512, 1, 1);
  cbr(s, u1, dn1.c2, u2, stats, zp, B, 512, 32, 32, 512, 3, 2);           // 16x16
  cbr(s, u2, dn1.c3, s2, stats, zp, B, 512, 16, 16, 512, 1, 1);
  conv_nchw(s, coef1, dn1.dwt_w, dn1.dwt_b, dc1, zp, B, 256, 32, 32, 128, 3, 1, 1);
  dwt_k<<<cdiv((size_t)B * 128 * 16 * 16, 256), 256, 0, s>>>(dc1, coef2, B, 128, 32, 32);

  cbr(s, coef2, dn2.c1, v1, stats, zp, B, 512, 16, 16, 1024, 1, 1);
  cbr(s, v1, dn2.c2, v2, stats, zp, B, 1024, 16, 16, 1024, 3, 2);         // 8x8 (c3 unused)
  conv_nchw(s, coef2, dn2.dwt_w, dn2.dwt_b, dc2, zp, B, 512, 16, 16, 256, 3, 1, 1);
  dwt_k<<<cdiv((size_t)B * 256 * 8 * 8, 256), 256, 0, s>>>(dc2, coef3, B, 256, 16, 16);

  // ---- ViT ----
  const int Mt = B * 64;
  gemm(s, coef3, embed_w, embed_b, pos, tokA, zp, Mt, 1024, 1024, 0, 1024,
       /*a_mode=*/2, 0, 0, /*res=pos*/2, 0, cd0);
  float* cur = tokA; float* nxt = tokB;
  for (int i = 0; i < 8; ++i) {
    layernorm_k<<<Mt, 256, 0, s>>>(cur, hbuf, blk[i].ln1g, blk[i].ln1b, 1024);
    gemm(s, hbuf, blk[i].wqkv, blk[i].bqkv, nullptr, qkvb, zp, Mt, 3072, 1024,
         1024, 3072, 0, 0, 0, 0, 0, cd0);
    attn_k<<<B * 4, 256, 0, s>>>(qkvb, obuf);
    gemm(s, obuf, blk[i].wo, blk[i].bo, cur, nxt, zp, Mt, 1024, 1024,
         1024, 1024, 0, 0, 0, /*res*/1, 0, cd0);
    { float* t = cur; cur = nxt; nxt = t; }
    layernorm_k<<<Mt, 256, 0, s>>>(cur, hbuf, blk[i].ln2g, blk[i].ln2b, 1024);
    gemm(s, hbuf, blk[i].w1, blk[i].b1, nullptr, m1b, zp, Mt, 512, 1024,
         1024, 512, 0, 0, 0, 0, /*gelu*/2, cd0);
    gemm(s, m1b, blk[i].w2, blk[i].b2, cur, nxt, zp, Mt, 1024, 512,
         512, 1024, 0, 0, 0, /*res*/1, 0, cd0);
    { float* t = cur; cur = nxt; nxt = t; }
  }
  layernorm_k<<<Mt, 256, 0, s>>>(cur, hbuf, lnfg, lnfb, 1024);
  tok2nchw_k<<<cdiv((size_t)Mt * 1024, 256), 256, 0, s>>>(hbuf, xn, B, 64, 1024);

  cbr(s, xn, bottle, bot, stats, zp, B, 1024, 8, 8, 512, 3, 1);           // (B,512,8,8)

  // ---- decoder ----
  cbr(s, coef3, up0.dwt_up, hp0, stats, zp, B, 1024, 8, 8, 512, 3, 1);
  idwt_k<<<cdiv((size_t)B * 512 * 64, 256), 256, 0, s>>>(bot, hp0, cat0, B, 512, 8, 8, 1024);
  copych_k<<<cdiv((size_t)B * 512 * 256, 256), 256, 0, s>>>(s2, cat0, B, 512, 256, 1024, 512);
  cbr(s, cat0, up0.conv, i1, stats, zp, B, 1024, 16, 16, 256, 3, 1);

  cbr(s, coef2, up1.dwt_up, hp1, stats, zp, B, 512, 16, 16, 256, 3, 1);
  idwt_k<<<cdiv((size_t)B * 256 * 256, 256), 256, 0, s>>>(i1, hp1, cat1, B, 256, 16, 16, 512);
  copych_k<<<cdiv((size_t)B * 256 * 1024, 256), 256, 0, s>>>(s1, cat1, B, 256, 1024, 512, 256);
  cbr(s, cat1, up1.conv, i2, stats, zp, B, 512, 32, 32, 128, 3, 1);

  cbr(s, coef1, up2.dwt_up, hp2, stats, zp, B, 256, 32, 32, 128, 3, 1);
  idwt_k<<<cdiv((size_t)B * 128 * 1024, 256), 256, 0, s>>>(i2, hp2, cat2, B, 128, 32, 32, 256);
  copych_k<<<cdiv((size_t)B * 128 * 4096, 256), 256, 0, s>>>(s0, cat2, B, 128, 4096, 256, 128);
  cbr(s, cat2, up2.conv, i3, stats, zp, B, 256, 64, 64, 64, 3, 1);

  idwt_k<<<cdiv((size_t)B * 64 * 4096, 256), 256, 0, s>>>(i3, nullptr, d4in, B, 64, 64, 64, 64);
  cbr(s, d4in, up3_last, d4, stats, zp, B, 64, 128, 128, 16, 3, 1);

  outconv_k<<<cdiv((size_t)B * 16384, 256), 256, 0, s>>>(d4, out_w, out_b,
                                                         (float*)d_out, B, 16, 16384);
}
